// TransformerBlock_60756607369640
// MI455X (gfx1250) — matrix operand order repaired and checked by reading, not compiled, not run on hardware
//
#include <hip/hip_runtime.h>
#include <cstdint>

// ---------------------------------------------------------------------------
// Transformer block for MI455X (gfx1250, wave32, WMMA + async-to-LDS).
// All GEMMs use v_wmma_f32_16x16x32_bf16 (bf16 inputs, fp32 accumulate).
// Weights are pre-transposed to [N][K] bf16 once, so every GEMM/attention
// tile is staged into LDS purely with GLOBAL_LOAD_ASYNC_TO_LDS_B128 (ASYNCcnt)
// and the inner loop is only ds_load_b128 fragment gathers + WMMAs.
// ---------------------------------------------------------------------------

typedef __attribute__((ext_vector_type(16))) __bf16 v16bf;
typedef __attribute__((ext_vector_type(8)))  float  v8f;

#define D_MODEL 2048
#define N_KV    8
#define D_HEAD  64
#define D_KV    (N_KV * D_HEAD)   // 512
#define D_MLP   8192
#define BATCH   2
#define SEQ     2048
#define ROWS    (BATCH * SEQ)     // 4096

__device__ __forceinline__ uint16_t f2bf(float f) {
    uint32_t u = __builtin_bit_cast(uint32_t, f);
    u += 0x7FFFu + ((u >> 16) & 1u);           // round-to-nearest-even
    return (uint16_t)(u >> 16);
}

// Load a 16-element bf16 fragment as two 16B chunks (p 16B-aligned, gap in elems).
__device__ __forceinline__ v16bf ld_frag(const uint16_t* p, int gap) {
    uint4 t[2];
    t[0] = *(const uint4*)(p);
    t[1] = *(const uint4*)(p + gap);
    v16bf r;
    __builtin_memcpy(&r, t, 32);
    return r;
}

__device__ __forceinline__ v8f vzero8() {
    v8f z;
    for (int i = 0; i < 8; ++i) z[i] = 0.0f;
    return z;
}

// Async global->LDS 16B copy (CDNA5 GLOBAL_LOAD_ASYNC_TO_LDS_B128, GVS mode).
__device__ __forceinline__ void async_ld_b128(uint32_t lds_off, uint32_t voff,
                                              uint64_t base) {
    asm volatile("global_load_async_to_lds_b128 %0, %1, %2"
                 :: "v"(lds_off), "v"(voff), "s"(base)
                 : "memory");
}
__device__ __forceinline__ void wait_async0() {
    asm volatile("s_wait_asynccnt 0" ::: "memory");
}
__device__ __forceinline__ uint32_t lds_off_of(const void* p) {
    // Generic pointers to LDS carry the LDS byte offset in the low 32 bits.
    return (uint32_t)(uintptr_t)p;
}

// ---------------------------------------------------------------------------
// Weight preparation (all GEMM B operands stored transposed: [N][K] bf16)
// ---------------------------------------------------------------------------

// W_Q (32, 2048, 64) -> Wq_eff^T bf16 [512][2048]: sum over each group of 4
// heads (reference einsum sums q over the group dim -> fold into the weight).
__global__ void k_wqeff_t(const float* __restrict__ WQ, uint16_t* __restrict__ dst) {
    int i = blockIdx.x * 256 + threadIdx.x;   // n = 512*2048
    int n = i >> 11, d = i & 2047;
    int kv = n >> 6, e = n & 63;
    float s = 0.f;
    for (int g = 0; g < 4; ++g)
        s += WQ[((size_t)(kv * 4 + g) * D_MODEL + d) * D_HEAD + e];
    dst[i] = f2bf(s);
}

// W_K / W_V (8, 2048, 64) -> W^T bf16 [512][2048]
__global__ void k_wkv_t(const float* __restrict__ W, uint16_t* __restrict__ dst) {
    int i = blockIdx.x * 256 + threadIdx.x;   // n = 512*2048
    int n = i >> 11, d = i & 2047;
    int kv = n >> 6, e = n & 63;
    dst[i] = f2bf(W[((size_t)kv * D_MODEL + d) * D_HEAD + e]);
}

// Tiled fp32->bf16 transpose: src [R][C] f32 -> dst [C][R] bf16.
// 32x32 tiles through LDS; both global sides coalesced.
__global__ __launch_bounds__(256) void k_cvt_t(const float* __restrict__ src,
                                               uint16_t* __restrict__ dst,
                                               int R, int C) {
    __shared__ uint16_t t[32][34];
    int tx = threadIdx.x & 31, ty = threadIdx.x >> 5;   // 32 x 8
    int c0 = blockIdx.x * 32, r0 = blockIdx.y * 32;
    for (int i = 0; i < 4; ++i) {
        int r = ty + i * 8;
        t[r][tx] = f2bf(src[(size_t)(r0 + r) * C + c0 + tx]);
    }
    __syncthreads();
    for (int i = 0; i < 4; ++i) {
        int r = ty + i * 8;
        dst[(size_t)(c0 + r) * R + r0 + tx] = t[tx][r];
    }
}

// ---------------------------------------------------------------------------
// rmsnorm (faithful: x / sqrt(var(x)+eps) * g + b, centered var)
// ---------------------------------------------------------------------------
__global__ __launch_bounds__(256) void k_rmsnorm(const float* __restrict__ x,
                                                 const float* __restrict__ g,
                                                 const float* __restrict__ bb,
                                                 uint16_t* __restrict__ out) {
    int row = blockIdx.x, tid = threadIdx.x;
    const float* xr = x + (size_t)row * D_MODEL;
    float s = 0.f, s2 = 0.f;
    for (int i = tid; i < D_MODEL; i += 256) {
        float v = xr[i];
        s += v; s2 += v * v;
    }
    for (int m = 1; m < 32; m <<= 1) {
        s  += __shfl_xor(s,  m, 32);
        s2 += __shfl_xor(s2, m, 32);
    }
    __shared__ float sh[2][8];
    int w = tid >> 5, l = tid & 31;
    if (l == 0) { sh[0][w] = s; sh[1][w] = s2; }
    __syncthreads();
    float ts = 0.f, ts2 = 0.f;
    for (int i = 0; i < 8; ++i) { ts += sh[0][i]; ts2 += sh[1][i]; }
    float mu  = ts  * (1.0f / D_MODEL);
    float var = ts2 * (1.0f / D_MODEL) - mu * mu;
    float inv = 1.0f / sqrtf(var + 1e-5f);
    for (int i = tid; i < D_MODEL; i += 256)
        out[(size_t)row * D_MODEL + i] = f2bf(xr[i] * inv * g[i] + bb[i]);
}

// ---------------------------------------------------------------------------
// Tiled bf16 WMMA GEMM:  C[M,N] = A[M,K] @ B[K,N], with B given TRANSPOSED
// as Bt[N][K] bf16. WG tile 128x128, 8 waves, wave tile 32x64, K-step 32,
// double-buffered LDS. Both tiles staged with async global->LDS b128, so the
// inner loop is: 4 async DMAs (next tile) + 12 ds_load_b128 + 8 WMMAs.
// Epilogue: *scale, +bias[col], +resid[row,col], relu; store f32, bf16, or
// transposed bf16 (outBt[col][row], used to feed attention V directly).
// ---------------------------------------------------------------------------
__global__ __launch_bounds__(256) void k_gemm(const uint16_t* __restrict__ A,
                                              const uint16_t* __restrict__ Bt,
                                              int M, int N, int K, float scale,
                                              const float* __restrict__ bias,
                                              const float* __restrict__ resid,
                                              int relu,
                                              float* __restrict__ outF,
                                              uint16_t* __restrict__ outB,
                                              uint16_t* __restrict__ outBt) {
    __shared__ alignas(16) uint16_t As[2][128 * 40];   // [row][k], row stride 80B
    __shared__ alignas(16) uint16_t Bs[2][128 * 40];   // [col][k]
    const int tid = threadIdx.x;
    const int w = tid >> 5, l = tid & 31;
    const int hi = l >> 4, lo = l & 15;
    const int row0 = blockIdx.y * 128, col0 = blockIdx.x * 128;
    const int wm = (w & 3) * 32, wn = (w >> 2) * 64;

    // Per-thread async chunk coordinates: 128 rows x 4 chunks of 16B, 2 iters.
    const int cr0 = tid >> 2, cc = tid & 3;
    const int cr1 = cr0 + 64;

    v8f acc[2][4];
    for (int i = 0; i < 2; ++i)
        for (int j = 0; j < 4; ++j) acc[i][j] = vzero8();

    auto stage = [&](int buf, int k0) {
        uint64_t baseA = (uint64_t)(uintptr_t)(A  + (size_t)row0 * K + k0);
        uint64_t baseB = (uint64_t)(uintptr_t)(Bt + (size_t)col0 * K + k0);
        async_ld_b128(lds_off_of(&As[buf][cr0 * 40 + cc * 8]),
                      (uint32_t)(cr0 * K * 2 + cc * 16), baseA);
        async_ld_b128(lds_off_of(&As[buf][cr1 * 40 + cc * 8]),
                      (uint32_t)(cr1 * K * 2 + cc * 16), baseA);
        async_ld_b128(lds_off_of(&Bs[buf][cr0 * 40 + cc * 8]),
                      (uint32_t)(cr0 * K * 2 + cc * 16), baseB);
        async_ld_b128(lds_off_of(&Bs[buf][cr1 * 40 + cc * 8]),
                      (uint32_t)(cr1 * K * 2 + cc * 16), baseB);
    };

    stage(0, 0);
    wait_async0();
    __syncthreads();

    int p = 0;
    for (int k0 = 0; k0 < K; k0 += 32, p ^= 1) {
        const int kn = k0 + 32;
        const bool more = kn < K;
        if (more) stage(p ^ 1, kn);   // next tile in flight during compute

        v16bf af[2], bfrag[4];
        for (int i = 0; i < 2; ++i) {
            const uint16_t* pA = &As[p][(wm + i * 16 + lo) * 40 + hi * 8];
            af[i] = ld_frag(pA, 16);
        }
        for (int j = 0; j < 4; ++j) {
            const uint16_t* pB = &Bs[p][(wn + j * 16 + lo) * 40 + hi * 8];
            bfrag[j] = ld_frag(pB, 16);
        }
        for (int i = 0; i < 2; ++i)
            for (int j = 0; j < 4; ++j)
                acc[i][j] = __builtin_amdgcn_wmma_f32_16x16x32_bf16(
                    false, af[i], false, bfrag[j], (short)0, acc[i][j], false, false);

        if (more) wait_async0();
        __syncthreads();
    }

    // Epilogue. C layout: elem e of lane l -> row e+8*hi, col lo.
    for (int i = 0; i < 2; ++i)
        for (int j = 0; j < 4; ++j)
            for (int e = 0; e < 8; ++e) {
                int rg = row0 + wm + i * 16 + e + 8 * hi;
                int cg = col0 + wn + j * 16 + lo;
                float v = acc[i][j][e] * scale;
                if (bias)  v += bias[cg];
                if (resid) v += resid[(size_t)rg * N + cg];
                if (relu)  v = fmaxf(v, 0.0f);
                if (outF)       outF[(size_t)rg * N + cg] = v;
                else if (outB)  outB[(size_t)rg * N + cg] = f2bf(v);
                else            outBt[(size_t)cg * M + rg] = f2bf(v);
            }
}

// ---------------------------------------------------------------------------
// Flash attention (causal, online softmax), 8 effective KV heads.
// Q pre-scaled by 1/sqrt(64). Grid: (S/64, N_KV, B). Block: 128 (4 waves),
// each wave owns 16 query rows; KV streamed in 64-key tiles through LDS.
// K tile from Kb[row][d], V tile from VBt[d][row] -- both staged purely with
// async global->LDS b128 (V was written transposed by its GEMM).
// ---------------------------------------------------------------------------
__global__ __launch_bounds__(128) void k_attn(const uint16_t* __restrict__ Qb,
                                              const uint16_t* __restrict__ Kb,
                                              const uint16_t* __restrict__ VBt,
                                              uint16_t* __restrict__ Zb) {
    __shared__ alignas(16) uint16_t Ks[64 * 72];       // [key][d]
    __shared__ alignas(16) uint16_t Vt[64 * 72];       // [d][key]
    __shared__ alignas(16) uint16_t Ps[4][16 * 72];    // per-wave P [qrow][key]
    const int tid = threadIdx.x;
    const int w = tid >> 5, l = tid & 31;
    const int hi = l >> 4, lo = l & 15;
    const int qt = blockIdx.x, kv = blockIdx.y, b = blockIdx.z;
    const int q0 = qt * 64, qw = q0 + w * 16;
    const size_t rowbase = (size_t)b * SEQ;

    // Q fragments straight from global (A-matrix layout, 16B loads).
    v16bf qf[2];
    {
        const uint16_t* qp = Qb + (rowbase + qw + lo) * D_KV + kv * 64;
        qf[0] = ld_frag(qp + hi * 8, 16);
        qf[1] = ld_frag(qp + 32 + hi * 8, 16);
    }

    float m_r[8], l_r[8];
    v8f accz[4];
    for (int e = 0; e < 8; ++e) { m_r[e] = -3.0e38f; l_r[e] = 0.0f; }
    for (int dt = 0; dt < 4; ++dt) accz[dt] = vzero8();

    const int ntiles = qt + 1;   // causal: only KV tiles at or below this q tile
    for (int t = 0; t < ntiles; ++t) {
        __syncthreads();
        // Async-stage K tile [64 keys][64 d] and V^T tile [64 d][64 keys]:
        // 512 16B chunks each; 4 per thread per matrix.
        {
            uint64_t baseK = (uint64_t)(uintptr_t)
                (Kb + (rowbase + t * 64) * D_KV + kv * 64);
            uint64_t baseV = (uint64_t)(uintptr_t)
                (VBt + (size_t)(kv * 64) * ROWS + rowbase + t * 64);
            for (int i = 0; i < 4; ++i) {
                int f = tid + i * 128;
                int r = f >> 3, c = f & 7;
                async_ld_b128(lds_off_of(&Ks[r * 72 + c * 8]),
                              (uint32_t)(r * D_KV * 2 + c * 16), baseK);
                async_ld_b128(lds_off_of(&Vt[r * 72 + c * 8]),
                              (uint32_t)(r * ROWS * 2 + c * 16), baseV);
            }
        }
        wait_async0();
        __syncthreads();

        // Scores: 4 key-tiles of 16, each = 2 chained WMMAs over d_head=64.
        v8f s[4];
        for (int nt = 0; nt < 4; ++nt) {
            v8f sa = vzero8();
            const uint16_t* kp = Ks + (nt * 16 + lo) * 72;   // B: lane = key col
            v16bf b0 = ld_frag(kp + hi * 8, 16);
            v16bf b1 = ld_frag(kp + 32 + hi * 8, 16);
            sa = __builtin_amdgcn_wmma_f32_16x16x32_bf16(false, qf[0], false, b0, (short)0, sa, false, false);
            sa = __builtin_amdgcn_wmma_f32_16x16x32_bf16(false, qf[1], false, b1, (short)0, sa, false, false);
            for (int e = 0; e < 8; ++e) {
                int qg = qw + e + 8 * hi;
                int kg = t * 64 + nt * 16 + lo;
                if (kg > qg) sa[e] = -100000.0f;   // reference IGNORE
            }
            s[nt] = sa;
        }

        // Online softmax row stats (rows live in 16-lane halves; xor<16 reduces).
        float mnew[8], alpha[8], psum[8];
        for (int e = 0; e < 8; ++e) {
            float mx = fmaxf(fmaxf(s[0][e], s[1][e]), fmaxf(s[2][e], s[3][e]));
            for (int msk = 1; msk < 16; msk <<= 1)
                mx = fmaxf(mx, __shfl_xor(mx, msk, 32));
            mnew[e]  = fmaxf(m_r[e], mx);
            alpha[e] = __expf(m_r[e] - mnew[e]);
            m_r[e]   = mnew[e];
            psum[e]  = 0.0f;
        }
        // p = exp(s - m); stage P into LDS (A-matrix source order).
        for (int nt = 0; nt < 4; ++nt)
            for (int e = 0; e < 8; ++e) {
                float p = __expf(s[nt][e] - mnew[e]);
                psum[e] += p;
                Ps[w][(e + 8 * hi) * 72 + nt * 16 + lo] = f2bf(p);
            }
        for (int e = 0; e < 8; ++e) {
            float ps = psum[e];
            for (int msk = 1; msk < 16; msk <<= 1)
                ps += __shfl_xor(ps, msk, 32);
            l_r[e] = alpha[e] * l_r[e] + ps;
        }
        for (int dt = 0; dt < 4; ++dt)
            for (int e = 0; e < 8; ++e)
                accz[dt][e] *= alpha[e];

        // Intra-wave RAW on Ps (DS ops are in-order per wave; wait to be safe).
        asm volatile("s_wait_dscnt 0" ::: "memory");

        // P fragments (A layout) and V fragments (B layout), then P @ V.
        v16bf pf[2];
        {
            const uint16_t* pp = Ps[w] + lo * 72;
            pf[0] = ld_frag(pp + hi * 8, 16);
            pf[1] = ld_frag(pp + 32 + hi * 8, 16);
        }
        for (int dt = 0; dt < 4; ++dt) {
            const uint16_t* vp = Vt + (dt * 16 + lo) * 72;
            v16bf v0 = ld_frag(vp + hi * 8, 16);
            v16bf v1 = ld_frag(vp + 32 + hi * 8, 16);
            accz[dt] = __builtin_amdgcn_wmma_f32_16x16x32_bf16(false, pf[0], false, v0, (short)0, accz[dt], false, false);
            accz[dt] = __builtin_amdgcn_wmma_f32_16x16x32_bf16(false, pf[1], false, v1, (short)0, accz[dt], false, false);
        }
    }

    // z = acc / l  -> bf16 [row, kv*64+d]
    for (int dt = 0; dt < 4; ++dt)
        for (int e = 0; e < 8; ++e) {
            float z = accz[dt][e] / l_r[e];
            int rg = qw + e + 8 * hi;
            Zb[(rowbase + rg) * D_KV + kv * 64 + dt * 16 + lo] = f2bf(z);
        }
}

// ---------------------------------------------------------------------------
// Host-side orchestration
// ---------------------------------------------------------------------------
extern "C" void kernel_launch(void* const* d_in, const int* in_sizes, int n_in,
                              void* d_out, int out_size, void* d_ws, size_t ws_size,
                              hipStream_t stream) {
    (void)in_sizes; (void)n_in; (void)out_size; (void)ws_size;
    const float* resid_pre = (const float*)d_in[0];
    const float* W_Q   = (const float*)d_in[1];
    const float* W_K   = (const float*)d_in[2];
    const float* W_V   = (const float*)d_in[3];
    const float* W_O   = (const float*)d_in[4];
    const float* g1    = (const float*)d_in[5];
    const float* b1    = (const float*)d_in[6];
    const float* g2    = (const float*)d_in[7];
    const float* b2    = (const float*)d_in[8];
    const float* W_in  = (const float*)d_in[9];
    const float* b_in  = (const float*)d_in[10];
    const float* W_out = (const float*)d_in[11];
    const float* b_out = (const float*)d_in[12];
    float* out = (float*)d_out;

    uint8_t* ws = (uint8_t*)d_ws;
    size_t o = 0;
    auto take = [&](size_t bytes) -> uint8_t* {
        uint8_t* p = ws + o;
        o = (o + bytes + 255) & ~(size_t)255;
        return p;
    };
    // Transposed bf16 weights: [N][K]
    uint16_t* WQET  = (uint16_t*)take((size_t)D_KV * D_MODEL * 2);    // [512][2048]
    uint16_t* WKT   = (uint16_t*)take((size_t)D_KV * D_MODEL * 2);
    uint16_t* WVT   = (uint16_t*)take((size_t)D_KV * D_MODEL * 2);
    uint16_t* WOT   = (uint16_t*)take((size_t)D_MODEL * D_KV * 2);    // [2048][512]
    uint16_t* WINT  = (uint16_t*)take((size_t)D_MLP * D_MODEL * 2);   // [8192][2048]
    uint16_t* WOUTT = (uint16_t*)take((size_t)D_MODEL * D_MLP * 2);   // [2048][8192]
    uint16_t* LN1   = (uint16_t*)take((size_t)ROWS * D_MODEL * 2);
    uint16_t* QB    = (uint16_t*)take((size_t)ROWS * D_KV * 2);
    uint16_t* KB    = (uint16_t*)take((size_t)ROWS * D_KV * 2);
    uint16_t* VBT   = (uint16_t*)take((size_t)D_KV * ROWS * 2);       // [512][4096]
    uint16_t* ZB    = (uint16_t*)take((size_t)ROWS * D_KV * 2);
    float*    RMID  = (float*)   take((size_t)ROWS * D_MODEL * 4);
    uint16_t* LN2   = (uint16_t*)take((size_t)ROWS * D_MODEL * 2);
    uint16_t* HB    = (uint16_t*)take((size_t)ROWS * D_MLP * 2);

    // Weight prep: bf16 + transpose (one-time, coalesced tiled transposes).
    {
        int n = D_KV * D_MODEL;
        k_wqeff_t<<<n / 256, 256, 0, stream>>>(W_Q, WQET);
        k_wkv_t  <<<n / 256, 256, 0, stream>>>(W_K, WKT);
        k_wkv_t  <<<n / 256, 256, 0, stream>>>(W_V, WVT);
        // W_O (8,64,2048) == [512][2048] -> WOT [2048][512]
        k_cvt_t<<<dim3(D_MODEL / 32, D_KV / 32), 256, 0, stream>>>(W_O, WOT, D_KV, D_MODEL);
        // W_in [2048][8192] -> WINT [8192][2048]
        k_cvt_t<<<dim3(D_MLP / 32, D_MODEL / 32), 256, 0, stream>>>(W_in, WINT, D_MODEL, D_MLP);
        // W_out [8192][2048] -> WOUTT [2048][8192]
        k_cvt_t<<<dim3(D_MODEL / 32, D_MLP / 32), 256, 0, stream>>>(W_out, WOUTT, D_MLP, D_MODEL);
    }

    // ln1 = rmsnorm(resid_pre)
    k_rmsnorm<<<ROWS, 256, 0, stream>>>(resid_pre, g1, b1, LN1);

    // Q (group-summed, pre-scaled 1/sqrt(d_head)), K, V (V stored transposed)
    dim3 gqkv(D_KV / 128, ROWS / 128);
    k_gemm<<<gqkv, 256, 0, stream>>>(LN1, WQET, ROWS, D_KV, D_MODEL, 0.125f,
                                     nullptr, nullptr, 0, nullptr, QB, nullptr);
    k_gemm<<<gqkv, 256, 0, stream>>>(LN1, WKT, ROWS, D_KV, D_MODEL, 1.0f,
                                     nullptr, nullptr, 0, nullptr, KB, nullptr);
    k_gemm<<<gqkv, 256, 0, stream>>>(LN1, WVT, ROWS, D_KV, D_MODEL, 1.0f,
                                     nullptr, nullptr, 0, nullptr, nullptr, VBT);

    // Flash attention
    k_attn<<<dim3(SEQ / 64, N_KV, BATCH), 128, 0, stream>>>(QB, KB, VBT, ZB);

    // resid_mid = Z @ W_O + resid_pre
    k_gemm<<<dim3(D_MODEL / 128, ROWS / 128), 256, 0, stream>>>(
        ZB, WOT, ROWS, D_MODEL, D_KV, 1.0f, nullptr, resid_pre, 0, RMID, nullptr, nullptr);

    // ln2 = rmsnorm(resid_mid)
    k_rmsnorm<<<ROWS, 256, 0, stream>>>(RMID, g2, b2, LN2);

    // H = relu(ln2 @ W_in + b_in)
    k_gemm<<<dim3(D_MLP / 128, ROWS / 128), 256, 0, stream>>>(
        LN2, WINT, ROWS, D_MLP, D_MODEL, 1.0f, b_in, nullptr, 1, nullptr, HB, nullptr);

    // out = H @ W_out + b_out + resid_mid
    k_gemm<<<dim3(D_MODEL / 128, ROWS / 128), 256, 0, stream>>>(
        HB, WOUTT, ROWS, D_MODEL, D_MLP, 1.0f, b_out, RMID, 0, out, nullptr, nullptr);
}
